// GlobalLocalTransformerBlock_2740189134917
// MI455X (gfx1250) — matrix operand order repaired and checked by reading, not compiled, not run on hardware
//
#include <hip/hip_runtime.h>
#include <hip/hip_bf16.h>
#include <math.h>

// ---------------------------------------------------------------------------
// PEGASUS-X global/local transformer block for MI455X (gfx1250, wave32).
// GEMMs: bf16 x bf16 -> f32 via v_wmma_f32_16x16x32_bf16, 128x128x32 tiles,
// double-buffered LDS with async global->LDS staging (ASYNCcnt).
// ---------------------------------------------------------------------------

typedef __bf16 bf16_t;
typedef __attribute__((ext_vector_type(16))) __bf16 v16bf;
typedef __attribute__((ext_vector_type(8)))  __bf16 v8bf;
typedef __attribute__((ext_vector_type(8)))  float  v8f;
typedef __attribute__((ext_vector_type(4)))  unsigned int v4u;

#define BB   4
#define TT   4096
#define DD   1024
#define GG   64
#define HH   16
#define HDD  64
#define BLKK 64
#define KBLK (TT / BLKK)
#define MLPD 4096
#define NEGV (-1e9f)

__device__ __forceinline__ float bf2f(bf16_t b) {
  unsigned short u = __builtin_bit_cast(unsigned short, b);
  unsigned int   x = ((unsigned int)u) << 16;
  return __builtin_bit_cast(float, x);
}
__device__ __forceinline__ bf16_t f2bf(float f) {
  unsigned int x = __builtin_bit_cast(unsigned int, f);
  unsigned int r = x + 0x7FFFu + ((x >> 16) & 1u);   // round-to-nearest-even
  unsigned short u = (unsigned short)(r >> 16);
  return __builtin_bit_cast(bf16_t, u);
}
__device__ __forceinline__ float gelu_tanh(float x) {
  float x3 = x * x * x;
  return 0.5f * x * (1.0f + tanhf(0.7978845608028654f * (x + 0.044715f * x3)));
}

// ------------------------------ f32 -> bf16 --------------------------------
__global__ void f32_to_bf16_kernel(const float* __restrict__ s,
                                   bf16_t* __restrict__ d, int n) {
  int i = blockIdx.x * blockDim.x + threadIdx.x;
  if (i < n) d[i] = f2bf(s[i]);
}

// ------------------------------ LayerNorm ----------------------------------
__global__ __launch_bounds__(256)
void layernorm_to_bf16_kernel(const float* __restrict__ x,
                              const float* __restrict__ sc,
                              const float* __restrict__ bi,
                              bf16_t* __restrict__ out) {
  __shared__ float red[256];
  __shared__ float red2[256];
  const int row = blockIdx.x;
  const float* xr = x + (size_t)row * DD;
  float s = 0.f, s2 = 0.f;
  for (int i = threadIdx.x; i < DD; i += 256) {
    float v = xr[i];
    s += v; s2 += v * v;
  }
  red[threadIdx.x] = s; red2[threadIdx.x] = s2;
  __syncthreads();
  for (int off = 128; off > 0; off >>= 1) {
    if ((int)threadIdx.x < off) {
      red[threadIdx.x]  += red[threadIdx.x + off];
      red2[threadIdx.x] += red2[threadIdx.x + off];
    }
    __syncthreads();
  }
  float mean = red[0] * (1.0f / DD);
  float var  = red2[0] * (1.0f / DD) - mean * mean;
  float inv  = rsqrtf(var + 1e-6f);
  bf16_t* orow = out + (size_t)row * DD;
  for (int i = threadIdx.x; i < DD; i += 256)
    orow[i] = f2bf((xr[i] - mean) * inv * sc[i] + bi[i]);
}

// ------------------------- WMMA tiled GEMM (bf16) --------------------------
// Tile: 128(M) x 128(N) x 32(K). 8 waves, each wave owns 16 rows x 128 cols.
// A tile staged via async global->LDS (b128); B tile transposed in VGPRs and
// stored with ds_store_b128. Double-buffered LDS.
#define MT 128
#define NT 128
#define KT 32
#define LDS_S (KT + 8)
#define NACC (NT / 16)

enum { EPI_BF16 = 0, EPI_BIAS_GELU_BF16 = 1, EPI_RES_F32 = 2 };

__device__ __forceinline__ v16bf ld_frag16(const bf16_t* p0, const bf16_t* p1) {
  v8bf lo = *(const v8bf*)p0;
  v8bf hi = *(const v8bf*)p1;
  v16bf r;
#pragma unroll
  for (int i = 0; i < 8; ++i) { r[i] = lo[i]; r[i + 8] = hi[i]; }
  return r;
}

template <int EPI>
__global__ __launch_bounds__(256)
void gemm_bf16_kernel(const bf16_t* __restrict__ A,
                      const bf16_t* __restrict__ Bw,
                      int M, int N, int K,
                      bf16_t* __restrict__ outB, float* __restrict__ outF,
                      const float* __restrict__ bias,
                      const float* __restrict__ res, float scale) {
  __shared__ bf16_t As[2][MT][LDS_S];   // row-major M x K tiles
  __shared__ bf16_t Bs[2][NT][LDS_S];   // transposed: Bs[n][k]
  const int tid    = threadIdx.x;
  const int lane   = tid & 31;
  const int wave   = tid >> 5;
  const int laneHi = lane >> 4;
  const int lane16 = lane & 15;
  const int blockM = blockIdx.y * MT;
  const int blockN = blockIdx.x * NT;

  // staging decomposition
  const int ar  = tid >> 2;            // A rows: ar and ar+64
  const int ac  = (tid & 3) * 8;       // A k-offset (8 bf16 = 16B)
  const int bn2 = (tid & 63) * 2;      // B column pair
  const int bkq = (tid >> 6) * 8;      // B k-octet

  auto stage = [&](int buf, int k0) {
    // ---- A tile: async global -> LDS, 2 x b128 per thread ----
#pragma unroll
    for (int it = 0; it < 2; ++it) {
      int r = ar + it * 64;
      unsigned lds = (unsigned)(uintptr_t)&As[buf][r][ac];
      const bf16_t* g = A + (size_t)(blockM + r) * K + k0 + ac;
      asm volatile("global_load_async_to_lds_b128 %0, %1, off"
                   :: "v"(lds), "v"((unsigned long long)(uintptr_t)g)
                   : "memory");
    }
    // ---- B tile: gather 8 k-rows for 2 columns, transpose-pack in VGPRs ----
    unsigned w[8];
#pragma unroll
    for (int dk = 0; dk < 8; ++dk)
      w[dk] = *(const unsigned*)(Bw + (size_t)(k0 + bkq + dk) * N + blockN + bn2);
    v4u c0, c1;
#pragma unroll
    for (int i = 0; i < 4; ++i) {
      c0[i] = (w[2 * i] & 0xFFFFu) | (w[2 * i + 1] << 16);
      c1[i] = (w[2 * i] >> 16)     | (w[2 * i + 1] & 0xFFFF0000u);
    }
    *(v4u*)&Bs[buf][bn2][bkq]     = c0;
    *(v4u*)&Bs[buf][bn2 + 1][bkq] = c1;
  };

  v8f acc[NACC];
#pragma unroll
  for (int t = 0; t < NACC; ++t)
#pragma unroll
    for (int i = 0; i < 8; ++i) acc[t][i] = 0.f;

  const int nK = K / KT;
  stage(0, 0);
  for (int kt = 0; kt < nK; ++kt) {
    const int buf = kt & 1;
    // drain this wave's async LDS writes, then workgroup barrier
    asm volatile("s_wait_asynccnt 0" ::: "memory");
    __syncthreads();
    if (kt + 1 < nK) stage(buf ^ 1, (kt + 1) * KT);

    // A fragment: lanes 0-15 K{0-7,16-23}, lanes 16-31 K{8-15,24-31}
    const int am  = wave * 16 + lane16;
    const int akb = laneHi * 8;
    v16bf af = ld_frag16(&As[buf][am][akb], &As[buf][am][akb + 16]);
#pragma unroll
    for (int t = 0; t < NACC; ++t) {
      const int bn = t * 16 + lane16;
      const int bk = laneHi * 8;         // lanes 0-15: K{0-7,16-23}, 16-31: K{8-15,24-31}
      v16bf bfr = ld_frag16(&Bs[buf][bn][bk], &Bs[buf][bn][bk + 16]);
      acc[t] = __builtin_amdgcn_wmma_f32_16x16x32_bf16(
          false, af, false, bfr, (short)0, acc[t], false, false);
    }
  }

  // Epilogue. C layout: VGPR r -> M = r + 8*laneHi, N = lane16.
#pragma unroll
  for (int t = 0; t < NACC; ++t) {
#pragma unroll
    for (int r = 0; r < 8; ++r) {
      int row = blockM + wave * 16 + r + 8 * laneHi;
      int col = blockN + t * 16 + lane16;
      float v = acc[t][r];
      if (EPI == EPI_BF16) {
        outB[(size_t)row * N + col] = f2bf(v * scale);
      } else if (EPI == EPI_BIAS_GELU_BF16) {
        outB[(size_t)row * N + col] = f2bf(gelu_tanh(v + bias[col]));
      } else {
        float b = bias ? bias[col] : 0.f;
        outF[(size_t)row * N + col] = v + b + res[(size_t)row * N + col];
      }
    }
  }
}

// --------------------------- local attention -------------------------------
// One WG per (b, block, h); 64 threads, each owns one query row.
__global__ __launch_bounds__(64)
void attn_local_kernel(const bf16_t* __restrict__ q, const bf16_t* __restrict__ k,
                       const bf16_t* __restrict__ v, const bf16_t* __restrict__ kg,
                       const bf16_t* __restrict__ vg,
                       const unsigned char* __restrict__ mask,
                       bf16_t* __restrict__ oloc) {
  __shared__ bf16_t ks[128][64];
  __shared__ bf16_t vs[128][64];
  __shared__ float  sc[64][128];
  const int f  = blockIdx.x;
  const int h  = f % HH;
  const int kb = (f / HH) % KBLK;
  const int b  = f / (HH * KBLK);
  const int t0 = kb * BLKK;
  const int t  = threadIdx.x;

  // keys/values: rows 0..63 global tokens, rows 64..127 local block
  {
    const bf16_t* s0 = kg + ((size_t)(b * GG + t)) * DD + h * HDD;
    const bf16_t* s1 = vg + ((size_t)(b * GG + t)) * DD + h * HDD;
    const bf16_t* s2 = k + ((size_t)(b * TT + t0 + t)) * DD + h * HDD;
    const bf16_t* s3 = v + ((size_t)(b * TT + t0 + t)) * DD + h * HDD;
#pragma unroll
    for (int i = 0; i < 8; ++i) {
      *(v8bf*)&ks[t][i * 8]      = *(const v8bf*)(s0 + i * 8);
      *(v8bf*)&vs[t][i * 8]      = *(const v8bf*)(s1 + i * 8);
      *(v8bf*)&ks[64 + t][i * 8] = *(const v8bf*)(s2 + i * 8);
      *(v8bf*)&vs[64 + t][i * 8] = *(const v8bf*)(s3 + i * 8);
    }
  }
  float qr[64];
  {
    const bf16_t* qrow = q + ((size_t)(b * TT + t0 + t)) * DD + h * HDD;
#pragma unroll
    for (int i = 0; i < 8; ++i) {
      v8bf x = *(const v8bf*)(qrow + i * 8);
#pragma unroll
      for (int j = 0; j < 8; ++j) qr[i * 8 + j] = bf2f(x[j]);
    }
  }
  __syncthreads();

  for (int j = 0; j < 128; ++j) {
    float s = 0.f;
#pragma unroll
    for (int d = 0; d < 64; ++d) s += qr[d] * bf2f(ks[j][d]);
    if (j >= 64 && !mask[(size_t)b * TT + t0 + (j - 64)]) s = NEGV;
    sc[t][j] = s;
  }
  float m = -3.4e38f;
  for (int j = 0; j < 128; ++j) m = fmaxf(m, sc[t][j]);
  float o[64];
#pragma unroll
  for (int d = 0; d < 64; ++d) o[d] = 0.f;
  float l = 0.f;
  for (int j = 0; j < 128; ++j) {
    float p = __expf(sc[t][j] - m);
    l += p;
#pragma unroll
    for (int d = 0; d < 64; ++d) o[d] += p * bf2f(vs[j][d]);
  }
  float inv = 1.f / l;
  bf16_t* orow = oloc + ((size_t)(b * TT + t0 + t)) * DD + h * HDD;
#pragma unroll
  for (int d = 0; d < 64; ++d) orow[d] = f2bf(o[d] * inv);
}

// --------------------------- global attention ------------------------------
// One WG per (b, h); 64 threads = 64 global query rows; flash-style online
// softmax over 64-token chunks (chunk 0 = global keys, then 64 local chunks).
__global__ __launch_bounds__(64)
void attn_global_kernel(const bf16_t* __restrict__ qg, const bf16_t* __restrict__ kg,
                        const bf16_t* __restrict__ vg, const bf16_t* __restrict__ k,
                        const bf16_t* __restrict__ v,
                        const unsigned char* __restrict__ mask,
                        bf16_t* __restrict__ oglb) {
  __shared__ bf16_t kc[64][64];
  __shared__ bf16_t vc[64][64];
  __shared__ float  sc[64][64];
  const int f = blockIdx.x;
  const int h = f % HH;
  const int b = f / HH;
  const int t = threadIdx.x;           // global row g

  float qr[64];
  {
    const bf16_t* qrow = qg + ((size_t)(b * GG + t)) * DD + h * HDD;
#pragma unroll
    for (int i = 0; i < 8; ++i) {
      v8bf x = *(const v8bf*)(qrow + i * 8);
#pragma unroll
      for (int j = 0; j < 8; ++j) qr[i * 8 + j] = bf2f(x[j]);
    }
  }
  float o[64];
#pragma unroll
  for (int d = 0; d < 64; ++d) o[d] = 0.f;
  float m = -3.4e38f, l = 0.f;

  for (int c = 0; c <= TT / 64; ++c) {
    const bf16_t *ksrc, *vsrc;
    if (c == 0) {
      ksrc = kg + ((size_t)(b * GG + t)) * DD + h * HDD;
      vsrc = vg + ((size_t)(b * GG + t)) * DD + h * HDD;
    } else {
      int tok = (c - 1) * 64 + t;
      ksrc = k + ((size_t)(b * TT + tok)) * DD + h * HDD;
      vsrc = v + ((size_t)(b * TT + tok)) * DD + h * HDD;
    }
#pragma unroll
    for (int i = 0; i < 8; ++i) {
      *(v8bf*)&kc[t][i * 8] = *(const v8bf*)(ksrc + i * 8);
      *(v8bf*)&vc[t][i * 8] = *(const v8bf*)(vsrc + i * 8);
    }
    __syncthreads();

    for (int j = 0; j < 64; ++j) {
      float s = 0.f;
#pragma unroll
      for (int d = 0; d < 64; ++d) s += qr[d] * bf2f(kc[j][d]);
      if (c > 0 && !mask[(size_t)b * TT + (c - 1) * 64 + j]) s = NEGV;
      sc[t][j] = s;
    }
    float cm = -3.4e38f;
    for (int j = 0; j < 64; ++j) cm = fmaxf(cm, sc[t][j]);
    float nm = fmaxf(m, cm);
    float r  = __expf(m - nm);
    l *= r;
#pragma unroll
    for (int d = 0; d < 64; ++d) o[d] *= r;
    for (int j = 0; j < 64; ++j) {
      float p = __expf(sc[t][j] - nm);
      l += p;
#pragma unroll
      for (int d = 0; d < 64; ++d) o[d] += p * bf2f(vc[j][d]);
    }
    m = nm;
    __syncthreads();
  }
  float inv = 1.f / l;
  bf16_t* orow = oglb + ((size_t)(b * GG + t)) * DD + h * HDD;
#pragma unroll
  for (int d = 0; d < 64; ++d) orow[d] = f2bf(o[d] * inv);
}

// ------------------------------- launcher ----------------------------------
extern "C" void kernel_launch(void* const* d_in, const int* in_sizes, int n_in,
                              void* d_out, int out_size, void* d_ws, size_t ws_size,
                              hipStream_t stream) {
  (void)in_sizes; (void)n_in; (void)out_size; (void)ws_size;
  const float* g_in   = (const float*)d_in[0];
  const float* x_in   = (const float*)d_in[1];
  const float* ln1_s  = (const float*)d_in[2];
  const float* ln1_b  = (const float*)d_in[3];
  const float* ln1g_s = (const float*)d_in[4];
  const float* ln1g_b = (const float*)d_in[5];
  const float* Wq  = (const float*)d_in[6];
  const float* Wk  = (const float*)d_in[7];
  const float* Wv  = (const float*)d_in[8];
  const float* Wgq = (const float*)d_in[9];
  const float* Wgk = (const float*)d_in[10];
  const float* Wgv = (const float*)d_in[11];
  const float* Wo  = (const float*)d_in[12];
  const float* Wgo = (const float*)d_in[13];
  const float* ln2_s = (const float*)d_in[14];
  const float* ln2_b = (const float*)d_in[15];
  const float* W1 = (const float*)d_in[16];
  const float* b1 = (const float*)d_in[17];
  const float* W2 = (const float*)d_in[18];
  const float* b2 = (const float*)d_in[19];
  const unsigned char* pmask = (const unsigned char*)d_in[20];

  float* out_global = (float*)d_out;                               // [B,G,D]
  float* out_local  = (float*)d_out + (size_t)BB * GG * DD;        // [B,T,D]

  char* ws = (char*)d_ws;
  size_t off = 0;
  auto alloc = [&](size_t bytes) -> size_t {
    size_t o = off; off += (bytes + 255) & ~(size_t)255; return o;
  };
  const size_t BT = (size_t)BB * TT;   // 16384 rows
  const size_t BG = (size_t)BB * GG;   // 256 rows

  size_t xln_o  = alloc(BT * DD * 2);
  size_t q_o    = alloc(BT * DD * 2);
  size_t k_o    = alloc(BT * DD * 2);
  size_t v_o    = alloc(BT * DD * 2);
  size_t gxln_o = alloc(BG * DD * 2);
  size_t qg_o   = alloc(BG * DD * 2);
  size_t kg_o   = alloc(BG * DD * 2);
  size_t vg_o   = alloc(BG * DD * 2);
  size_t oloc_o = alloc(BT * DD * 2);
  size_t oglb_o = alloc(BG * DD * 2);
  size_t x2_o   = alloc(BT * DD * 4);
  size_t gx2_o  = alloc(BG * DD * 4);
  size_t ln2x_o = alloc(BT * DD * 2);
  size_t ln2g_o = alloc(BG * DD * 2);
  size_t wq_o  = alloc((size_t)DD * DD * 2);
  size_t wk_o  = alloc((size_t)DD * DD * 2);
  size_t wv_o  = alloc((size_t)DD * DD * 2);
  size_t wgq_o = alloc((size_t)DD * DD * 2);
  size_t wgk_o = alloc((size_t)DD * DD * 2);
  size_t wgv_o = alloc((size_t)DD * DD * 2);
  size_t wo_o  = alloc((size_t)DD * DD * 2);
  size_t wgo_o = alloc((size_t)DD * DD * 2);
  size_t w1_o  = alloc((size_t)DD * MLPD * 2);
  size_t w2_o  = alloc((size_t)MLPD * DD * 2);
  // Aliased buffers (qkv/xln dead before MLP): h = 16384x4096 bf16 = exactly
  // xln+q+k+v; hg overlays qg..oloc (dead as well).
  size_t h_o  = xln_o;
  size_t hg_o = qg_o;

  auto bf = [&](size_t o) { return (bf16_t*)(ws + o); };

  // ---- weight conversions ----
  struct Cv { const float* s; size_t o; int n; };
  const Cv cvs[] = {
    {Wq, wq_o, DD * DD},  {Wk, wk_o, DD * DD},  {Wv, wv_o, DD * DD},
    {Wgq, wgq_o, DD * DD},{Wgk, wgk_o, DD * DD},{Wgv, wgv_o, DD * DD},
    {Wo, wo_o, DD * DD},  {Wgo, wgo_o, DD * DD},
    {W1, w1_o, DD * MLPD},{W2, w2_o, MLPD * DD},
  };
  for (const Cv& c : cvs)
    f32_to_bf16_kernel<<<(c.n + 255) / 256, 256, 0, stream>>>(c.s, bf(c.o), c.n);

  // ---- LayerNorm 1 ----
  layernorm_to_bf16_kernel<<<(int)BT, 256, 0, stream>>>(x_in, ln1_s, ln1_b, bf(xln_o));
  layernorm_to_bf16_kernel<<<(int)BG, 256, 0, stream>>>(g_in, ln1g_s, ln1g_b, bf(gxln_o));

  // ---- QKV projections (bf16 WMMA GEMMs) ----
  auto gemm_b = [&](size_t a, size_t w, size_t o, int M, int N, int K, float sc) {
    gemm_bf16_kernel<EPI_BF16><<<dim3(N / NT, M / MT), 256, 0, stream>>>(
        bf(a), bf(w), M, N, K, bf(o), nullptr, nullptr, nullptr, sc);
  };
  gemm_b(xln_o,  wq_o,  q_o,  (int)BT, DD, DD, 0.125f);   // 1/sqrt(64)
  gemm_b(xln_o,  wk_o,  k_o,  (int)BT, DD, DD, 1.f);
  gemm_b(xln_o,  wv_o,  v_o,  (int)BT, DD, DD, 1.f);
  gemm_b(gxln_o, wgq_o, qg_o, (int)BG, DD, DD, 0.125f);
  gemm_b(gxln_o, wgk_o, kg_o, (int)BG, DD, DD, 1.f);
  gemm_b(gxln_o, wgv_o, vg_o, (int)BG, DD, DD, 1.f);

  // ---- attention ----
  attn_local_kernel<<<BB * KBLK * HH, 64, 0, stream>>>(
      bf(q_o), bf(k_o), bf(v_o), bf(kg_o), bf(vg_o), pmask, bf(oloc_o));
  attn_global_kernel<<<BB * HH, 64, 0, stream>>>(
      bf(qg_o), bf(kg_o), bf(vg_o), bf(k_o), bf(v_o), pmask, bf(oglb_o));

  // ---- output projection + residual (f32) ----
  gemm_bf16_kernel<EPI_RES_F32><<<dim3(DD / NT, (int)BT / MT), 256, 0, stream>>>(
      bf(oloc_o), bf(wo_o), (int)BT, DD, DD, nullptr, (float*)(ws + x2_o),
      nullptr, x_in, 1.f);
  gemm_bf16_kernel<EPI_RES_F32><<<dim3(DD / NT, (int)BG / MT), 256, 0, stream>>>(
      bf(oglb_o), bf(wgo_o), (int)BG, DD, DD, nullptr, (float*)(ws + gx2_o),
      nullptr, g_in, 1.f);

  // ---- LayerNorm 2 (shared) ----
  layernorm_to_bf16_kernel<<<(int)BT, 256, 0, stream>>>(
      (const float*)(ws + x2_o), ln2_s, ln2_b, bf(ln2x_o));
  layernorm_to_bf16_kernel<<<(int)BG, 256, 0, stream>>>(
      (const float*)(ws + gx2_o), ln2_s, ln2_b, bf(ln2g_o));

  // ---- MLP1: gelu(x W1 + b1) -> bf16 ----
  gemm_bf16_kernel<EPI_BIAS_GELU_BF16><<<dim3(MLPD / NT, (int)BT / MT), 256, 0, stream>>>(
      bf(ln2x_o), bf(w1_o), (int)BT, MLPD, DD, bf(h_o), nullptr, b1, nullptr, 1.f);
  gemm_bf16_kernel<EPI_BIAS_GELU_BF16><<<dim3(MLPD / NT, (int)BG / MT), 256, 0, stream>>>(
      bf(ln2g_o), bf(w1_o), (int)BG, MLPD, DD, bf(hg_o), nullptr, b1, nullptr, 1.f);

  // ---- MLP2: h W2 + b2 + residual -> d_out (f32) ----
  gemm_bf16_kernel<EPI_RES_F32><<<dim3(DD / NT, (int)BT / MT), 256, 0, stream>>>(
      bf(h_o), bf(w2_o), (int)BT, DD, MLPD, nullptr, out_local, b2,
      (const float*)(ws + x2_o), 1.f);
  gemm_bf16_kernel<EPI_RES_F32><<<dim3(DD / NT, (int)BG / MT), 256, 0, stream>>>(
      bf(hg_o), bf(w2_o), (int)BG, DD, MLPD, nullptr, out_global, b2,
      (const float*)(ws + gx2_o), 1.f);
}
